// RNNDownBeatProc_39805756899674
// MI455X (gfx1250) — matrix operand order repaired and checked by reading, not compiled, not run on hardware
//
#include <hip/hip_runtime.h>
#include <hip/hip_bf16.h>
#include <math.h>

typedef __bf16 bf16_t;
typedef __attribute__((ext_vector_type(16))) __bf16 v16bf;
typedef __attribute__((ext_vector_type(8)))  __bf16 v8bf;
typedef __attribute__((ext_vector_type(8)))  float   v8f;

#define BATCH 64
#define TLEN  3000
#define FEAT  314
#define HID   25
#define G4    100          // 4*H gate width (i,f,g,o)
#define NDIR  112          // per-direction gate width padded to 7*16
#define NPACK 224          // both directions packed
#define NT    14           // N tiles = 224/16
#define NB    7            // N blocks of 2 tiles (32 cols)
#define KP0   320          // layer-0 K padded (314 -> 320, 10 k-tiles)
#define KP12  64           // layer-1/2 K padded (50 -> 64, 2 k-tiles)
#define MROWS (TLEN*BATCH) // 192000
#define MTILES (MROWS/16)  // 12000
#define MBLKS  (MTILES/2)  // 6000 (32-row blocks)

// ---- A fragment (16x32 bf16) from row-major bf16 [16 x ld], ISA 7.12.2 layout:
// lanes 0-15: row=lane, K halves {0..7,16..23}; lanes 16-31: row=lane-16, {8..15,24..31}
__device__ __forceinline__ v16bf load_a_frag(const bf16_t* base, int ld, int k0) {
  int lane = threadIdx.x & 31;
  const bf16_t* p = base + (size_t)(lane & 15) * ld + k0 + ((lane >> 4) << 3);
  v8bf lo = *(const v8bf*)p;
  v8bf hi = *(const v8bf*)(p + 16);
  v16bf r;
#pragma unroll
  for (int i = 0; i < 8; ++i) { r[i] = lo[i]; r[i + 8] = hi[i]; }
  return r;
}

// ============================ pre-pack kernels ==============================

// x (B,T,F) f32 -> Xbf (T*B, KP0) bf16, zero-padded cols
__global__ void pack_x(const float* __restrict__ x, bf16_t* __restrict__ X) {
  size_t idx = (size_t)blockIdx.x * blockDim.x + threadIdx.x;
  if (idx >= (size_t)MROWS * KP0) return;
  int col = (int)(idx % KP0);
  size_t row = idx / KP0;
  int t = (int)(row / BATCH), b = (int)(row % BATCH);
  float v = (col < FEAT) ? x[((size_t)b * TLEN + t) * FEAT + col] : 0.0f;
  X[idx] = (bf16_t)v;
}

// w_ih (2, 100, kreal) f32 -> B-fragment-linear bf16 [ktiles][NT][32 lanes][16 halves]
__global__ void pack_wih(const float* __restrict__ wih, int kreal, int ktiles,
                         bf16_t* __restrict__ frag) {
  int idx = blockIdx.x * blockDim.x + threadIdx.x;
  if (idx >= ktiles * NT * 512) return;
  int i    = idx & 15;
  int lane = (idx >> 4) & 31;
  int nt   = (idx >> 9) % NT;
  int kt   = idx / (NT * 512);
  int k = kt * 32 + ((i >> 3) << 4) + ((lane >> 4) << 3) + (i & 7);   // lanes 0-15: K {0..7,16..23}; 16-31: K {8..15,24..31}
  int c = nt * 16 + (lane & 15);
  int dd = c / NDIR, g = c % NDIR;
  float v = 0.0f;
  if (g < G4 && k < kreal) v = wih[((size_t)dd * G4 + g) * kreal + k];
  frag[idx] = (bf16_t)v;
}

// w_hh (2, 100, 25) f32 -> [2 dirs][7 nt][32][16] fragments of W_hh^T (K=25 pad 32)
__global__ void pack_whh(const float* __restrict__ whh, bf16_t* __restrict__ frag) {
  int idx = blockIdx.x * blockDim.x + threadIdx.x;
  if (idx >= 2 * 7 * 512) return;
  int i    = idx & 15;
  int lane = (idx >> 4) & 31;
  int nt   = (idx >> 9) % 7;
  int dd   = idx / (7 * 512);
  int k = ((i >> 3) << 4) + ((lane >> 4) << 3) + (i & 7);
  int c = nt * 16 + (lane & 15);
  float v = 0.0f;
  if (c < G4 && k < HID) v = whh[((size_t)dd * G4 + c) * HID + k]; // B[k][c] = Whh^T
  frag[idx] = (bf16_t)v;
}

__global__ void pack_bias(const float* __restrict__ bih, const float* __restrict__ bhh,
                          float* __restrict__ bias) {
  int c = blockIdx.x * blockDim.x + threadIdx.x;
  if (c >= NPACK) return;
  int dd = c / NDIR, g = c % NDIR;
  bias[c] = (g < G4) ? (bih[dd * G4 + g] + bhh[dd * G4 + g]) : 0.0f;
}

__global__ void zero_bf(bf16_t* __restrict__ p, size_t n) {
  size_t i = (size_t)blockIdx.x * blockDim.x + threadIdx.x;
  if (i < n) p[i] = (bf16_t)0.0f;
}

// ===================== phase A: xg = X * W^T + bias (WMMA) ==================
// 2x2 register blocking: each wave computes a 32x32 output block (4 WMMAs per
// k-tile from 8 b128 loads -> 2:1 vmem:wmma instead of 4:1).
__global__ void __launch_bounds__(256)
gemm_xw(const bf16_t* __restrict__ X, int kp, int ktiles,
        const bf16_t* __restrict__ bfrag, const float* __restrict__ bias,
        float* __restrict__ out, int blocks_total) {
  int wb = (blockIdx.x << 3) + (threadIdx.x >> 5);
  if (wb >= blocks_total) return;
  int mb = wb / NB, nb = wb % NB;
  int lane = threadIdx.x & 31;
  v8f acc00 = {}, acc01 = {}, acc10 = {}, acc11 = {};
  const bf16_t* arow0 = X + (size_t)(mb * 32) * kp;
  const bf16_t* arow1 = arow0 + (size_t)16 * kp;
  const bf16_t* bp0 = bfrag + ((size_t)(2 * nb) * 32 + lane) * 16;
  const bf16_t* bp1 = bp0 + 512;
  const size_t bstride = (size_t)NT * 512;
  for (int kt = 0; kt < ktiles; ++kt) {
    v16bf a0 = load_a_frag(arow0, kp, kt * 32);
    v16bf a1 = load_a_frag(arow1, kp, kt * 32);
    v16bf b0 = *(const v16bf*)(bp0 + (size_t)kt * bstride);
    v16bf b1 = *(const v16bf*)(bp1 + (size_t)kt * bstride);
    acc00 = __builtin_amdgcn_wmma_f32_16x16x32_bf16(false, a0, false, b0, (short)0, acc00, false, false);
    acc01 = __builtin_amdgcn_wmma_f32_16x16x32_bf16(false, a0, false, b1, (short)0, acc01, false, false);
    acc10 = __builtin_amdgcn_wmma_f32_16x16x32_bf16(false, a1, false, b0, (short)0, acc10, false, false);
    acc11 = __builtin_amdgcn_wmma_f32_16x16x32_bf16(false, a1, false, b1, (short)0, acc11, false, false);
  }
  int c0   = nb * 32 + (lane & 15);
  int row0 = mb * 32 + ((lane >> 4) << 3);  // C layout: lanes 0-15 rows 0-7, 16-31 rows 8-15
  float bi0 = bias[c0], bi1 = bias[c0 + 16];
#pragma unroll
  for (int r = 0; r < 8; ++r) {
    out[(size_t)(row0 + r) * NPACK + c0]            = acc00[r] + bi0;
    out[(size_t)(row0 + r) * NPACK + c0 + 16]       = acc01[r] + bi1;
    out[(size_t)(row0 + 16 + r) * NPACK + c0]       = acc10[r] + bi0;
    out[(size_t)(row0 + 16 + r) * NPACK + c0 + 16]  = acc11[r] + bi1;
  }
}

// =============== phase B: sequential LSTM recurrence (WMMA/step) ============
// grid: (4 batch-tiles, 2 directions); block: 7 waves (224 threads).
// Each wave: one 16x16 gate tile per step; W_hh^T fragment is loop-invariant.
__global__ void __launch_bounds__(224)
lstm_recur(const float* __restrict__ xg, const bf16_t* __restrict__ hhfrag,
           bf16_t* __restrict__ Y) {
  __shared__ __align__(16) bf16_t h_lds[16 * 32];
  __shared__ float c_lds[16 * 26];
  __shared__ float g_lds[16 * NDIR];
  const int tid = threadIdx.x;
  const int wave = tid >> 5, lane = tid & 31;
  const int bt = blockIdx.x;   // batch tile (16 rows)
  const int d  = blockIdx.y;   // direction

  for (int i = tid; i < 16 * 32; i += 224) h_lds[i] = (bf16_t)0.0f;
  for (int i = tid; i < 16 * 26; i += 224) c_lds[i] = 0.0f;
  __syncthreads();

  const v16bf bw = *(const v16bf*)(hhfrag + ((size_t)(d * 7 + wave) * 32 + lane) * 16);
  const int col = wave * 16 + (lane & 15);
  const int r0  = (lane >> 4) << 3;

  for (int s = 0; s < TLEN; ++s) {
    const int t = d ? (TLEN - 1 - s) : s;
    v16bf a = load_a_frag(h_lds, 32, 0);
    v8f acc = {};
    acc = __builtin_amdgcn_wmma_f32_16x16x32_bf16(false, a, false, bw, (short)0, acc,
                                                  false, false);
#pragma unroll
    for (int r = 0; r < 8; ++r) g_lds[(r0 + r) * NDIR + col] = acc[r];
    __syncthreads();

    const float* xrow = xg + ((size_t)t * BATCH + bt * 16) * NPACK + d * NDIR;
    for (int idx = tid; idx < 16 * HID; idx += 224) {
      int b = idx & 15, j = idx >> 4;
      const float* xp = xrow + (size_t)b * NPACK;
      float gi = g_lds[b * NDIR + j]      + xp[j];
      float gf = g_lds[b * NDIR + j + 25] + xp[j + 25];
      float gg = g_lds[b * NDIR + j + 50] + xp[j + 50];
      float go = g_lds[b * NDIR + j + 75] + xp[j + 75];
      float si = 1.0f / (1.0f + __expf(-gi));
      float sf = 1.0f / (1.0f + __expf(-gf));
      float so = 1.0f / (1.0f + __expf(-go));
      float c = sf * c_lds[b * 26 + j] + si * tanhf(gg);
      float h = so * tanhf(c);
      c_lds[b * 26 + j] = c;
      h_lds[b * 32 + j] = (bf16_t)h;
      Y[((size_t)t * BATCH + bt * 16 + b) * 64 + d * HID + j] = (bf16_t)h;
    }
    __syncthreads();
  }
}

// ================================ FC head ===================================
__global__ void fc_kernel(const bf16_t* __restrict__ Y, const float* __restrict__ w,
                          const float* __restrict__ bias, float* __restrict__ out) {
  int row = blockIdx.x * blockDim.x + threadIdx.x;  // row = t*64 + b
  if (row >= MROWS) return;
  int t = row / BATCH, b = row % BATCH;
  const bf16_t* y = Y + (size_t)row * 64;
  float a0 = bias[0], a1 = bias[1], a2 = bias[2];
  for (int j = 0; j < 2 * HID; ++j) {
    float v = (float)y[j];
    a0 += v * w[j];
    a1 += v * w[2 * HID + j];
    a2 += v * w[4 * HID + j];
  }
  float* o = out + ((size_t)b * TLEN + t) * 3;
  o[0] = a0; o[1] = a1; o[2] = a2;
}

// ================================ launcher ==================================
extern "C" void kernel_launch(void* const* d_in, const int* in_sizes, int n_in,
                              void* d_out, int out_size, void* d_ws, size_t ws_size,
                              hipStream_t stream) {
  (void)in_sizes; (void)n_in; (void)out_size; (void)ws_size;
  const float* x = (const float*)d_in[0];
  const float* wih[3] = {(const float*)d_in[1], (const float*)d_in[5], (const float*)d_in[9]};
  const float* whh[3] = {(const float*)d_in[2], (const float*)d_in[6], (const float*)d_in[10]};
  const float* bih[3] = {(const float*)d_in[3], (const float*)d_in[7], (const float*)d_in[11]};
  const float* bhh[3] = {(const float*)d_in[4], (const float*)d_in[8], (const float*)d_in[12]};
  const float* fcw = (const float*)d_in[13];
  const float* fcb = (const float*)d_in[14];
  float* out = (float*)d_out;

  char* ws = (char*)d_ws;
  size_t off = 0;
  auto alloc = [&](size_t bytes) -> char* {
    char* p = ws + off;
    off = (off + bytes + 255) & ~(size_t)255;
    return p;
  };
  bf16_t* Xbf = (bf16_t*)alloc((size_t)MROWS * KP0 * sizeof(bf16_t));
  float*  xg  = (float*) alloc((size_t)MROWS * NPACK * sizeof(float));
  bf16_t* Ya  = (bf16_t*)alloc((size_t)MROWS * 64 * sizeof(bf16_t));
  bf16_t* Yb  = (bf16_t*)alloc((size_t)MROWS * 64 * sizeof(bf16_t));
  bf16_t* fih0 = (bf16_t*)alloc((size_t)10 * NT * 512 * sizeof(bf16_t));
  bf16_t* fih1 = (bf16_t*)alloc((size_t)2 * NT * 512 * sizeof(bf16_t));
  bf16_t* fih2 = (bf16_t*)alloc((size_t)2 * NT * 512 * sizeof(bf16_t));
  bf16_t* fhh[3];
  for (int l = 0; l < 3; ++l) fhh[l] = (bf16_t*)alloc((size_t)2 * 7 * 512 * sizeof(bf16_t));
  float* bias[3];
  for (int l = 0; l < 3; ++l) bias[l] = (float*)alloc(NPACK * sizeof(float));

  // ---- pre-pack
  {
    size_t total = (size_t)MROWS * KP0;
    pack_x<<<(unsigned)((total + 255) / 256), 256, 0, stream>>>(x, Xbf);
  }
  pack_wih<<<(10 * NT * 512 + 255) / 256, 256, 0, stream>>>(wih[0], FEAT, 10, fih0);
  pack_wih<<<(2 * NT * 512 + 255) / 256, 256, 0, stream>>>(wih[1], 2 * HID, 2, fih1);
  pack_wih<<<(2 * NT * 512 + 255) / 256, 256, 0, stream>>>(wih[2], 2 * HID, 2, fih2);
  for (int l = 0; l < 3; ++l) {
    pack_whh<<<(2 * 7 * 512 + 255) / 256, 256, 0, stream>>>(whh[l], fhh[l]);
    pack_bias<<<1, NPACK, 0, stream>>>(bih[l], bhh[l], bias[l]);
  }
  {
    size_t n = (size_t)MROWS * 64;
    zero_bf<<<(unsigned)((n + 255) / 256), 256, 0, stream>>>(Ya, n);
    zero_bf<<<(unsigned)((n + 255) / 256), 256, 0, stream>>>(Yb, n);
  }

  const int blocks_total = MBLKS * NB;          // 42000 wave-blocks
  const int gblocks = (blocks_total + 7) / 8;   // 8 waves per 256-thread block
  // layer 0
  gemm_xw<<<gblocks, 256, 0, stream>>>(Xbf, KP0, 10, fih0, bias[0], xg, blocks_total);
  lstm_recur<<<dim3(4, 2), 224, 0, stream>>>(xg, fhh[0], Ya);
  // layer 1
  gemm_xw<<<gblocks, 256, 0, stream>>>(Ya, KP12, 2, fih1, bias[1], xg, blocks_total);
  lstm_recur<<<dim3(4, 2), 224, 0, stream>>>(xg, fhh[1], Yb);
  // layer 2
  gemm_xw<<<gblocks, 256, 0, stream>>>(Yb, KP12, 2, fih2, bias[2], xg, blocks_total);
  lstm_recur<<<dim3(4, 2), 224, 0, stream>>>(xg, fhh[2], Ya);
  // FC head
  fc_kernel<<<(MROWS + 255) / 256, 256, 0, stream>>>(Ya, fcw, fcb, out);
}
